// LowAggreCell_25314537243097
// MI455X (gfx1250) — matrix operand order repaired and checked by reading, not compiled, not run on hardware
//
#include <hip/hip_runtime.h>
#include <math.h>

// ---------------------------------------------------------------------------
typedef __attribute__((ext_vector_type(16))) _Float16 f16x16;
typedef __attribute__((ext_vector_type(8)))  float    f32x8;

#define HH 128
#define WW 128
#define NB 8
#define NC 64
#define HW (HH*WW)               // 16384
#define PLANE_SMALL 4096         // 64*64
#define P_ELEMS (NB*HW)          // 131072 (pow2)
#define Q_ELEMS (NB*NC*HW)       // 8388608 (pow2)
#define BNS 0.9999950000374997f  // 1/sqrt(1+1e-5)

// ---------------------------------------------------------------------------
// Elementwise kernels
// ---------------------------------------------------------------------------
__device__ __forceinline__ float bilin_up2(const float* p, int id) {
  int yo = (id >> 7) & 127, xo = id & 127;
  const float sc = 63.0f / 127.0f;
  float fy = yo * sc; int y0 = (int)fy; float wy = fy - y0; int y1 = min(y0 + 1, 63);
  float fx = xo * sc; int x0 = (int)fx; float wx = fx - x0; int x1 = min(x0 + 1, 63);
  float v00 = p[y0 * 64 + x0], v01 = p[y0 * 64 + x1];
  float v10 = p[y1 * 64 + x0], v11 = p[y1 * 64 + x1];
  return (v00 * (1.f - wy) + v10 * wy) * (1.f - wx) +
         (v01 * (1.f - wy) + v11 * wy) * wx;
}

// two (8,1,64,64) maps -> x2 upsample + sigmoid -> two (8,128,128)
__global__ __launch_bounds__(256) void k_up_sigmoid2(const float* __restrict__ in0,
                                                     const float* __restrict__ in1,
                                                     float* __restrict__ out0,
                                                     float* __restrict__ out1) {
  int idx = blockIdx.x * 256 + threadIdx.x;
  if (idx >= 2 * P_ELEMS) return;
  const float* in = (idx < P_ELEMS) ? in0 : in1;
  float* out      = (idx < P_ELEMS) ? out0 : out1;
  int id = idx & (P_ELEMS - 1);
  int b = id >> 14;
  float v = bilin_up2(in + b * PLANE_SMALL, id & (HW - 1));
  out[id] = 1.f / (1.f + __expf(-v));
}

// two (8,64,64,64) tensors -> x2 upsample -> two (8,64,128,128)
__global__ __launch_bounds__(256) void k_up_c2(const float* __restrict__ in0,
                                               const float* __restrict__ in1,
                                               float* __restrict__ out0,
                                               float* __restrict__ out1) {
  long long idx = (long long)blockIdx.x * 256 + threadIdx.x;
  if (idx >= 2LL * Q_ELEMS) return;
  const float* in = (idx < Q_ELEMS) ? in0 : in1;
  float* out      = (idx < Q_ELEMS) ? out0 : out1;
  int id = (int)(idx & (Q_ELEMS - 1));
  int plane = id >> 14;
  out[id] = bilin_up2(in + plane * PLANE_SMALL, id & (HW - 1));
}

__global__ __launch_bounds__(256) void k_maxpool5(const float* __restrict__ in,
                                                  float* __restrict__ out) {
  int idx = blockIdx.x * 256 + threadIdx.x;
  if (idx >= P_ELEMS) return;
  int b = idx >> 14;
  int rem = idx & (HW - 1);
  int y = rem >> 7, x = rem & 127;
  const float* p = in + b * HW;
  float m = -1e30f;
  int y0 = max(y - 2, 0), y1 = min(y + 2, HH - 1);
  int x0 = max(x - 2, 0), x1 = min(x + 2, WW - 1);
  for (int yy = y0; yy <= y1; ++yy)
    for (int xx = x0; xx <= x1; ++xx)
      m = fmaxf(m, p[yy * WW + xx]);
  out[idx] = m;
}

__global__ __launch_bounds__(256) void k_halves(const float* __restrict__ f1,
                                                const float* __restrict__ f2,
                                                const float* __restrict__ p3s,
                                                const float* __restrict__ dil,
                                                float* __restrict__ ta, float* __restrict__ tb,
                                                float* __restrict__ tc, float* __restrict__ td) {
  int idx = blockIdx.x * 256 + threadIdx.x;
  if (idx >= Q_ELEMS) return;
  int s = idx & (HW - 1);
  int b = idx >> 20;
  float p  = p3s[(b << 14) + s];
  float pe = dil[(b << 14) + s] - p;
  float a = f1[idx], c = f2[idx];
  ta[idx] = a * (1.f + p);
  tb[idx] = c * (1.f + p);
  tc[idx] = a * (1.f + pe);
  td[idx] = c * (1.f + pe);
}

__global__ __launch_bounds__(256) void k_edge_scale(const float* __restrict__ e3s,
                                                    const float* __restrict__ f,
                                                    float* __restrict__ out) {
  int idx = blockIdx.x * 256 + threadIdx.x;
  if (idx >= Q_ELEMS) return;
  int s = idx & (HW - 1);
  int b = idx >> 20;
  out[idx] = f[idx] * (1.f + e3s[(b << 14) + s]);
}

// ---------------------------------------------------------------------------
// Weight prep: f16, gamma-folded, exact WMMA A-fragment register order:
//   dst[segoff + ((chunk*4 + ctile)*32 + lane)*16 + j]
// ---------------------------------------------------------------------------
#define NSEG 19
struct WPrep {
  const float* w[NSEG];
  const float* g[NSEG];
  int cin[NSEG];
  int ksz[NSEG];
  int end[NSEG + 1];
};

__global__ __launch_bounds__(256) void k_wprep(WPrep p, _Float16* __restrict__ dst,
                                               int total) {
  int e = blockIdx.x * 256 + threadIdx.x;
  if (e >= total) return;
  int s = 0;
  while (s < NSEG - 1 && e >= p.end[s + 1]) ++s;
  int le = e - p.end[s];
  int cin = p.cin[s], ksz = p.ksz[s];
  int ksq = ksz * ksz, cpt = cin >> 5;
  int j    = le & 15;
  int lane = (le >> 4) & 31;
  int ct   = (le >> 9) & 3;
  int c    = le >> 11;
  int tap = c / cpt;
  int cib = (c - tap * cpt) << 5;
  int half = lane >> 4, rr = lane & 15;
  int co = ct * 16 + rr;
  int kk = (j < 8) ? (half * 8 + j) : (16 + half * 8 + (j - 8));
  float v = p.w[s][(co * cin + cib + kk) * ksq + tap] * (p.g[s][co] * BNS);
  dst[e] = (_Float16)v;
}

// ---------------------------------------------------------------------------
// Fused implicit-GEMM conv with a halo-padded LDS row cache.
// Phases = NT terms x (CIN/64) halves; each phase stages 64 channels of
// NROWS input rows (+1-col zero halo for 3x3) once, then runs every tap /
// channel slice out of LDS. B fragments are software-pipelined one deep.
//   KSZ==3: block = 1 output row  (M=128), NROWS=3, 4 acc/wave, LDS 56KB
//   KSZ==1: block = 2 output rows (M=256), NROWS=2, 8 acc/wave, LDS 37KB
// ---------------------------------------------------------------------------
struct ConvArgs {
  const float* x0[3];     // per-term input (C=64)
  const float* x1;        // second 64ch of concat (term 0 only, CIN=128)
  const _Float16* wq;     // prepped weight base
  int woff[3];            // per-term segment offsets (f16 elements)
  const float* bt[3];     // per-term beta
  const float* res;
  float* out;
};

#define RS 72               // f16 per (row,col) cell stride: 144B, bank-safe

template <int KSZ, int CIN, int NT, bool RIN, bool ROUT>
__global__ __launch_bounds__(256) void k_conv(ConvArgs a) {
  constexpr int CPT = CIN / 32;              // chunks per tap (weights)
  constexpr int KSQ = KSZ * KSZ;
  constexpr int CPH = CIN / 64;              // channel halves
  constexpr int NROWS = (KSZ == 1) ? 2 : 3;  // staged input rows
  constexpr int NTIL = (KSZ == 1) ? 8 : 4;   // spatial tiles per wave
  constexpr int NPACK = NROWS * 128 * 8;     // 8-f16 packs per phase

  __shared__ __align__(16) _Float16 ldsX[NROWS * 130 * RS];

  const int tid  = threadIdx.x;
  const int lane = tid & 31;
  const int wave = tid >> 5;
  const int half = lane >> 4;
  const int r    = lane & 15;

  const int T = blockIdx.x;
  int b, ybase;
  if (KSZ == 1) { b = T >> 6; ybase = (T & 63) << 1; }   // 512 blocks, 2 rows
  else          { b = T >> 7; ybase = T & 127;       }   // 1024 blocks, 1 row

  const int ct  = wave & 3;                 // co tile (16 wide)
  const int st0 = (wave >> 2) * NTIL;       // spatial tiles per wave

  f32x8 acc[NTIL] = {};

  for (int p = 0; p < NT * CPH; ++p) {
    int t, chalf;
    if (CPH == 1) { t = p; chalf = 0; } else { t = p >> 1; chalf = p & 1; }
    if (p) __syncthreads();                 // previous phase done reading LDS

    // ---- stage: NROWS rows x 128 cols x 64 channels, packed b128 stores
    const float* src = (CIN == 128 && chalf == 1) ? a.x1 : a.x0[t];
#pragma unroll
    for (int i = 0; i < NPACK / 256; ++i) {
      int pp = tid + (i << 8);
      int m  = pp & 127;                    // x
      int q  = pp >> 7;
      int cg = q & 7;                       // channel group (8 ch)
      int rrow = q >> 3;                    // staged row index
      int yy = (KSZ == 1) ? (ybase + rrow) : (ybase + rrow - 1);
      bool ok = (unsigned)yy < (unsigned)HH;
      int rowidx = ((b * NC + cg * 8) * HH + yy) * WW + m;
      union { _Float16 h[8]; uint4 u; } pkt;
#pragma unroll
      for (int j = 0; j < 8; ++j) {
        float v = 0.f;
        if (ok) v = src[rowidx + j * HW];
        if (RIN) v = fmaxf(v, 0.f);
        pkt.h[j] = (_Float16)v;
      }
      *reinterpret_cast<uint4*>(&ldsX[(rrow * 130 + m + 1) * RS + cg * 8]) = pkt.u;
    }
    if (KSZ == 3) {                         // zero halo columns (xcol 0, 129)
      if (tid < NROWS * 2 * 8) {
        int cg = tid & 7;
        int pos = tid >> 3;
        int rrow = pos >> 1;
        int xcol = (pos & 1) ? 129 : 0;
        uint4 z = {0, 0, 0, 0};
        *reinterpret_cast<uint4*>(&ldsX[(rrow * 130 + xcol) * RS + cg * 8]) = z;
      }
    }
    __syncthreads();

    // ---- compute: all taps x 2 channel slices from the row cache
#pragma unroll 2
    for (int cc = 0; cc < KSQ * 2; ++cc) {
      int tap = cc >> 1, s = cc & 1;
      int kh = tap / KSZ, kw = tap - kh * KSZ;
      int cl = tap * CPT + chalf * 2 + s;   // weight chunk within term

      f16x16 af;
      {
        const _Float16* wp = a.wq + a.woff[t] + ((cl * 4 + ct) * 32 + lane) * 16;
        uint4* d = reinterpret_cast<uint4*>(&af);
        d[0] = *reinterpret_cast<const uint4*>(wp);
        d[1] = *reinterpret_cast<const uint4*>(wp + 8);
      }
      auto ldb = [&](int j) {
        int m = (st0 + j) * 16 + r;
        int rrow = (KSZ == 1) ? (m >> 7) : kh;
        int xcol = (KSZ == 1) ? ((m & 127) + 1) : (m + kw);
        f16x16 bf;
        const _Float16* xp = &ldsX[(rrow * 130 + xcol) * RS + s * 32 + half * 8];
        uint4* d = reinterpret_cast<uint4*>(&bf);
        d[0] = *reinterpret_cast<const uint4*>(xp);
        d[1] = *reinterpret_cast<const uint4*>(xp + 16);
        return bf;
      };
      // B fragments pipelined one deep
      f16x16 bc = ldb(0);
#pragma unroll
      for (int j = 0; j < NTIL; ++j) {
        f16x16 bn = bc;
        if (j + 1 < NTIL) bn = ldb(j + 1);
        acc[j] = __builtin_amdgcn_wmma_f32_16x16x32_f16(
            false, af, false, bc, (short)0, acc[j], false, false);
        bc = bn;
      }
    }
  }

  // ---- epilogue: + sum(beta) (+ReLU) (+residual)
#pragma unroll
  for (int j = 0; j < NTIL; ++j) {
    int m = (st0 + j) * 16 + r;
    int y = (KSZ == 1) ? (ybase + (m >> 7)) : ybase;
    int x = (KSZ == 1) ? (m & 127) : m;
#pragma unroll
    for (int v = 0; v < 8; ++v) {
      int co = ct * 16 + half * 8 + v;
      float bias = a.bt[0][co];
      if constexpr (NT > 1) bias += a.bt[1][co];
      if constexpr (NT > 2) bias += a.bt[2][co];
      float val = acc[j][v] + bias;
      if (ROUT) val = fmaxf(val, 0.f);
      int idx = ((b * NC + co) * HH + y) * WW + x;
      if (a.res) val += a.res[idx];
      a.out[idx] = val;
    }
  }
}

// ---------------------------------------------------------------------------
extern "C" void kernel_launch(void* const* d_in, const int* in_sizes, int n_in,
                              void* d_out, int out_size, void* d_ws, size_t ws_size,
                              hipStream_t stream) {
  const float* f1     = (const float*)d_in[0];
  const float* f2     = (const float*)d_in[1];
  const float* f3     = (const float*)d_in[2];
  const float* p3     = (const float*)d_in[3];
  const float* edge3  = (const float*)d_in[4];
  const float* fedge3 = (const float*)d_in[5];
  // params flattened: f1f2 @7, f12 @10, f12e @13, seg[i] @16+3i, edge[i] @34+3i,
  // agg[i] @46+3i, seg_low @52, edge_low @55, agg_low @58, seg_out @61, edge_out @64

  const size_t P = (size_t)P_ELEMS;
  const size_t Q = (size_t)Q_ELEMS;
  float* wsf = (float*)d_ws;
  float* p3s = wsf;
  float* e3s = wsf + P;
  float* tmp = wsf + 2 * P;
  float* dil = wsf + 3 * P;
  float* slot0 = wsf + 4 * P;
  auto S = [&](int i) { return slot0 + (size_t)i * Q; };
  // slots: 0 f3u  1 fe3u  2 ta/s3  3 tb/s4  4 tc/e2  5 td/e3
  //        6 f1f2/seg_low  7 f1f2_edge/agg_out  8 f12/edge_low  9 f12e/a2
  _Float16* wq = (_Float16*)(slot0 + (size_t)10 * Q);

  const int gP = (int)((P + 255) / 256);
  const int gQ = (int)((Q + 255) / 256);

  // ---- weight prep descriptor table
  const int swidx[NSEG] = {7, 10, 13, 52, 55, 58,
                           16, 19, 22, 25, 28, 31,
                           34, 37, 40, 43, 46, 61, 64};
  const int scin[NSEG]  = {128, 128, 128, 128, 128, 64,
                           64, 64, 64, 64, 64, 64,
                           64, 64, 64, 64, 64, 128, 128};
  const int sksz[NSEG]  = {1, 1, 1, 1, 1, 1,
                           3, 3, 3, 3, 3, 3,
                           3, 3, 3, 3, 3, 3, 3};
  WPrep wp{};
  int segoff[NSEG];
  int cum = 0;
  wp.end[0] = 0;
  for (int s = 0; s < NSEG; ++s) {
    wp.w[s] = (const float*)d_in[swidx[s]];
    wp.g[s] = (const float*)d_in[swidx[s] + 1];
    wp.cin[s] = scin[s];
    wp.ksz[s] = sksz[s];
    segoff[s] = cum;
    cum += (sksz[s] * sksz[s] * scin[s] / 32) * 2048;  // chunks * 2048 f16
    wp.end[s + 1] = cum;
  }
  k_wprep<<<dim3((cum + 255) / 256), dim3(256), 0, stream>>>(wp, wq, cum);

  auto bt = [&](int widx) { return (const float*)d_in[widx + 2]; };
  auto mk = [&](const float* x0, const float* x1, int seg, int widx, float* out,
                const float* res) {
    ConvArgs a{};
    a.x0[0] = x0; a.x1 = x1;
    a.wq = wq; a.woff[0] = segoff[seg]; a.bt[0] = bt(widx);
    a.res = res; a.out = out;
    return a;
  };
  auto term = [&](ConvArgs& a, int t, const float* x0, int seg, int widx) {
    a.x0[t] = x0; a.woff[t] = segoff[seg]; a.bt[t] = bt(widx);
  };

  // ---- preprocess
  k_up_sigmoid2<<<2 * gP, 256, 0, stream>>>(p3, edge3, p3s, e3s);
  k_maxpool5   <<<gP, 256, 0, stream>>>(p3s, tmp);
  k_maxpool5   <<<gP, 256, 0, stream>>>(tmp, dil);
  k_up_c2      <<<2 * gQ, 256, 0, stream>>>(f3, fedge3, S(0), S(1));  // f3u, fe3u
  k_halves     <<<gQ, 256, 0, stream>>>(f1, f2, p3s, dil, S(2), S(3), S(4), S(5));

  const dim3 G1(512), G3(1024), Blk(256);

  // ---- 1x1 preprocess convs (cbr)
  k_conv<1,128,1,false,true><<<G1, Blk, 0, stream>>>(mk(f1,   f2,   0, 7,  S(6), nullptr)); // f1f2
  k_edge_scale<<<gQ, 256, 0, stream>>>(e3s, S(6), S(7));                                    // f1f2_edge
  k_conv<1,128,1,false,true><<<G1, Blk, 0, stream>>>(mk(S(2), S(3), 1, 10, S(8), nullptr)); // f12
  k_conv<1,128,1,false,true><<<G1, Blk, 0, stream>>>(mk(S(4), S(5), 2, 13, S(9), nullptr)); // f12e

  // ---- cell step 0: s3 = rcb(f12)+rcb(f12e)+rcb(f3u) -> S2 (one fused GEMM)
  {
    ConvArgs a = mk(S(8), nullptr, 6, 16, S(2), nullptr);
    term(a, 1, S(9), 7, 19);
    term(a, 2, S(0), 8, 22);
    k_conv<3,64,3,true,false><<<G3, Blk, 0, stream>>>(a);
  }
  //              e2 = rcb(f1f2_edge)+rcb(f3u) -> S4
  {
    ConvArgs a = mk(S(7), nullptr, 12, 34, S(4), nullptr);
    term(a, 1, S(0), 13, 37);
    k_conv<3,64,2,true,false><<<G3, Blk, 0, stream>>>(a);
  }
  // ---- cell step 1: s4 = rcb(f12e)+rcb(f3u)+rcb(s3) -> S3
  {
    ConvArgs a = mk(S(9), nullptr, 9, 25, S(3), nullptr);
    term(a, 1, S(0), 10, 28);
    term(a, 2, S(2), 11, 31);
    k_conv<3,64,3,true,false><<<G3, Blk, 0, stream>>>(a);
  }
  //              e3 = rcb(f3u)+rcb(e2) -> S5
  {
    ConvArgs a = mk(S(0), nullptr, 14, 40, S(5), nullptr);
    term(a, 1, S(4), 15, 43);
    k_conv<3,64,2,true,false><<<G3, Blk, 0, stream>>>(a);
  }

  // ---- heads
  k_conv<1,128,1,false,true><<<G1, Blk, 0, stream>>>(mk(S(2), S(3), 3, 52, S(6), nullptr)); // seg_out
  k_conv<1,128,1,false,true><<<G1, Blk, 0, stream>>>(mk(S(4), S(5), 4, 55, S(8), nullptr)); // edge_out

  // ---- aggregation (source bug: agg[0] weights for BOTH terms) -> S9
  {
    ConvArgs a = mk(S(6), nullptr, 16, 46, S(9), nullptr);
    term(a, 1, S(8), 16, 46);
    k_conv<3,64,2,true,false><<<G3, Blk, 0, stream>>>(a);
  }
  k_conv<1,64,1,false,true><<<G1, Blk, 0, stream>>>(mk(S(9), nullptr, 5, 58, S(7), nullptr)); // agg_out

  // ---- finals (cbr 3x3 + residual) straight into d_out: [seg | edge]
  float* segf  = (float*)d_out;
  float* edgef = segf + Q;
  k_conv<3,128,1,false,true><<<G3, Blk, 0, stream>>>(mk(S(7), S(8), 18, 64, edgef, S(1))); // edge_final
  k_conv<3,128,1,false,true><<<G3, Blk, 0, stream>>>(mk(S(7), S(6), 17, 61, segf,  S(0))); // seg_final
}
